// UVNetGraphEncoder_72834055406398
// MI455X (gfx1250) — matrix operand order repaired and checked by reading, not compiled, not run on hardware
//
#include <hip/hip_runtime.h>

typedef __bf16 bf16_t;
typedef bf16_t v16bf __attribute__((ext_vector_type(16)));
typedef float  v8f   __attribute__((ext_vector_type(8)));

#define N_NODES 8192
#define N_EDGES 32768
#define HD 64
#define OUTD 128
#define NCHUNK 65               // 64 outer-product chunks + 1 bias chunk
#define CHUNK_ELEMS 4096        // 8 fragments * 512 bf16 elems per chunk

// ---------------------------------------------------------------------------
// Pre-pack W_edge (+ b_edge) into WMMA-B fragment-major bf16 layout.
// GEMM: msg[e,o] = sum_{K=(i,k)} P[e,K] * B[K,o],  B[(i,k),o] = W_edge[i*64+o, k]
// Bias chunk (c==64): B[i',o] = b_edge[i'*64+o].
// Flat index: (((c*2+kk)*4 + t)*32 + lane)*16 + j
//   lane<16 holds n=lane, k = kk*32 + j ; lane>=16 holds n=lane-16, k = kk*32 + 16 + j
// ---------------------------------------------------------------------------
__global__ __launch_bounds__(256) void pack_B_kernel(const float* __restrict__ We,
                                                     const float* __restrict__ be,
                                                     bf16_t* __restrict__ Bp)
{
    int idx = blockIdx.x * 256 + threadIdx.x;
    if (idx >= NCHUNK * CHUNK_ELEMS) return;
    const int j    = idx & 15;
    const int lane = (idx >> 4) & 31;
    const int t    = (idx >> 9) & 3;
    const int kk   = (idx >> 11) & 1;
    const int c    = idx >> 12;
    const int hi   = lane >> 4;
    const int n    = lane & 15;
    const int krow = kk * 32 + j + 16 * hi;      // k within 64-wide chunk
    float v;
    if (c < 64) v = We[(size_t)(c * 64 + t * 16 + n) * 64 + krow];
    else        v = be[krow * 64 + t * 16 + n];
    Bp[idx] = (bf16_t)v;
}

// ---------------------------------------------------------------------------
// Fused edge kernel: per 16-edge tile, msg = (x_src ⊗ ef | x_src) @ Bpacked,
// accumulated with v_wmma_f32_16x16x32_bf16, then atomically scattered to
// aggr[dst]. 4 waves / block, one tile per wave. Bias chunk peeled out of the
// hot loop; hot loop unrolled 2x so A-fragment register sets ping-pong and
// cover the WMMA->VALU WAR hazard window.
// ---------------------------------------------------------------------------
__global__ __launch_bounds__(128) void edge_msg_kernel(
    const float* __restrict__ ef, const float* __restrict__ x,
    const bf16_t* __restrict__ Bp, const int* __restrict__ src,
    const int* __restrict__ dst, float* __restrict__ aggr)
{
    __shared__ float ef_lds[4][16][65];   // +1 pad: conflict-free fragment gather
    __shared__ float xs_lds[4][16][65];
    __shared__ int   dst_lds[4][16];

    const int wave = threadIdx.x >> 5;
    const int lane = threadIdx.x & 31;
    const int e0   = (blockIdx.x * 4 + wave) * 16;

    if (lane < 16) dst_lds[wave][lane] = dst[e0 + lane];
    for (int r = 0; r < 16; ++r) {
        const int e = e0 + r;
        const int s = src[e];
        float2 ev = *(const float2*)(ef + (size_t)e * HD + lane * 2);
        float2 xv = *(const float2*)(x  + (size_t)s * HD + lane * 2);
        ef_lds[wave][r][lane * 2]     = ev.x;
        ef_lds[wave][r][lane * 2 + 1] = ev.y;
        xs_lds[wave][r][lane * 2]     = xv.x;
        xs_lds[wave][r][lane * 2 + 1] = xv.y;
    }
    __syncthreads();

    const int m  = lane & 15;   // edge row within tile (A-matrix M)
    const int hi = lane >> 4;

    // Gather this lane's ef values in WMMA A-fragment order (both K halves),
    // and pre-build the (loop-invariant) bias-chunk A fragments = x_src itself.
    float efk[32];
    v16bf ab0, ab1;
#pragma unroll
    for (int j = 0; j < 16; ++j) {
        const int kA = hi * 16 + j;
        efk[j]      = ef_lds[wave][m][kA];
        efk[16 + j] = ef_lds[wave][m][32 + kA];
        ab0[j] = (bf16_t)xs_lds[wave][m][kA];
        ab1[j] = (bf16_t)xs_lds[wave][m][32 + kA];
    }

    v8f c0 = {}, c1 = {}, c2 = {}, c3 = {};

#pragma unroll 2
    for (int chunk = 0; chunk < 64; ++chunk) {
        // A chunk = ef row scaled by per-lane scalar x_src[m, chunk]
        const float s = xs_lds[wave][m][chunk];
        v16bf a0, a1;
#pragma unroll
        for (int j = 0; j < 16; ++j) {
            a0[j] = (bf16_t)(efk[j] * s);
            a1[j] = (bf16_t)(efk[16 + j] * s);
        }

        const bf16_t* bb = Bp + (size_t)chunk * CHUNK_ELEMS + lane * 16;
        __builtin_prefetch(bb + CHUNK_ELEMS, 0, 1);
        v16bf b00 = *(const v16bf*)(bb + 0 * 512);
        v16bf b01 = *(const v16bf*)(bb + 1 * 512);
        v16bf b02 = *(const v16bf*)(bb + 2 * 512);
        v16bf b03 = *(const v16bf*)(bb + 3 * 512);
        v16bf b10 = *(const v16bf*)(bb + 4 * 512);
        v16bf b11 = *(const v16bf*)(bb + 5 * 512);
        v16bf b12 = *(const v16bf*)(bb + 6 * 512);
        v16bf b13 = *(const v16bf*)(bb + 7 * 512);

        c0 = __builtin_amdgcn_wmma_f32_16x16x32_bf16(false, a0, false, b00, (short)0, c0, false, false);
        c1 = __builtin_amdgcn_wmma_f32_16x16x32_bf16(false, a0, false, b01, (short)0, c1, false, false);
        c2 = __builtin_amdgcn_wmma_f32_16x16x32_bf16(false, a0, false, b02, (short)0, c2, false, false);
        c3 = __builtin_amdgcn_wmma_f32_16x16x32_bf16(false, a0, false, b03, (short)0, c3, false, false);
        c0 = __builtin_amdgcn_wmma_f32_16x16x32_bf16(false, a1, false, b10, (short)0, c0, false, false);
        c1 = __builtin_amdgcn_wmma_f32_16x16x32_bf16(false, a1, false, b11, (short)0, c1, false, false);
        c2 = __builtin_amdgcn_wmma_f32_16x16x32_bf16(false, a1, false, b12, (short)0, c2, false, false);
        c3 = __builtin_amdgcn_wmma_f32_16x16x32_bf16(false, a1, false, b13, (short)0, c3, false, false);
    }

    // Peeled bias chunk (c == 64): A = x_src, B = packed b_edge.
    {
        const bf16_t* bb = Bp + (size_t)64 * CHUNK_ELEMS + lane * 16;
        v16bf b00 = *(const v16bf*)(bb + 0 * 512);
        v16bf b01 = *(const v16bf*)(bb + 1 * 512);
        v16bf b02 = *(const v16bf*)(bb + 2 * 512);
        v16bf b03 = *(const v16bf*)(bb + 3 * 512);
        v16bf b10 = *(const v16bf*)(bb + 4 * 512);
        v16bf b11 = *(const v16bf*)(bb + 5 * 512);
        v16bf b12 = *(const v16bf*)(bb + 6 * 512);
        v16bf b13 = *(const v16bf*)(bb + 7 * 512);

        c0 = __builtin_amdgcn_wmma_f32_16x16x32_bf16(false, ab0, false, b00, (short)0, c0, false, false);
        c1 = __builtin_amdgcn_wmma_f32_16x16x32_bf16(false, ab0, false, b01, (short)0, c1, false, false);
        c2 = __builtin_amdgcn_wmma_f32_16x16x32_bf16(false, ab0, false, b02, (short)0, c2, false, false);
        c3 = __builtin_amdgcn_wmma_f32_16x16x32_bf16(false, ab0, false, b03, (short)0, c3, false, false);
        c0 = __builtin_amdgcn_wmma_f32_16x16x32_bf16(false, ab1, false, b10, (short)0, c0, false, false);
        c1 = __builtin_amdgcn_wmma_f32_16x16x32_bf16(false, ab1, false, b11, (short)0, c1, false, false);
        c2 = __builtin_amdgcn_wmma_f32_16x16x32_bf16(false, ab1, false, b12, (short)0, c2, false, false);
        c3 = __builtin_amdgcn_wmma_f32_16x16x32_bf16(false, ab1, false, b13, (short)0, c3, false, false);
    }

    // C/D layout: VGPR v, lanes<16 -> (M=v, N=lane); lanes>=16 -> (M=8+v, N=lane-16)
#pragma unroll
    for (int v = 0; v < 8; ++v) {
        const int row = v + 8 * hi;
        const int d   = dst_lds[wave][row];
        float* ap = aggr + (size_t)d * HD + m;
        atomicAdd(ap +  0, c0[v]);
        atomicAdd(ap + 16, c1[v]);
        atomicAdd(ap + 32, c2[v]);
        atomicAdd(ap + 48, c3[v]);
    }
}

// ---------------------------------------------------------------------------
// h = aggr + x @ W_root + b_root ; accumulate per-channel sum / sumsq for BN.
// ---------------------------------------------------------------------------
__global__ __launch_bounds__(256) void node_root_stats_kernel(
    const float* __restrict__ aggr, const float* __restrict__ x,
    const float* __restrict__ Wr, const float* __restrict__ br,
    float* __restrict__ h, float* __restrict__ sum, float* __restrict__ sumsq)
{
    __shared__ float sW[64 * 64];
    __shared__ float ssum[64], ssq[64];
    for (int i = threadIdx.x; i < 64 * 64; i += 256) sW[i] = Wr[i];
    if (threadIdx.x < 64) { ssum[threadIdx.x] = 0.f; ssq[threadIdx.x] = 0.f; }
    __syncthreads();

    const int idx = blockIdx.x * 256 + threadIdx.x;
    const int n = idx >> 6, o = idx & 63;
    const float* xr = x + (size_t)n * 64;
    float acc = br[o];
#pragma unroll
    for (int i = 0; i < 64; ++i) acc += xr[i] * sW[i * 64 + o];
    acc += aggr[idx];
    h[idx] = acc;
    atomicAdd(&ssum[o], acc);
    atomicAdd(&ssq[o], acc * acc);
    __syncthreads();
    if (threadIdx.x < 64) {
        atomicAdd(&sum[threadIdx.x],   ssum[threadIdx.x]);
        atomicAdd(&sumsq[threadIdx.x], ssq[threadIdx.x]);
    }
}

__global__ void bn_finalize_kernel(const float* __restrict__ sum,
                                   const float* __restrict__ sumsq,
                                   const float* __restrict__ gamma,
                                   const float* __restrict__ beta,
                                   float* __restrict__ sc, float* __restrict__ sh)
{
    const int o = threadIdx.x;
    const float inv_n = 1.0f / (float)N_NODES;
    const float mean  = sum[o] * inv_n;
    const float var   = sumsq[o] * inv_n - mean * mean;
    const float s     = gamma[o] * rsqrtf(var + 1e-5f);
    sc[o] = s;
    sh[o] = beta[o] - mean * s;
}

__global__ __launch_bounds__(256) void bn_apply_kernel(
    const float* __restrict__ h, const float* __restrict__ sc,
    const float* __restrict__ sh, float* __restrict__ out)
{
    const int idx = blockIdx.x * 256 + threadIdx.x;
    const int o = idx & 63;
    float v = h[idx] * sc[o] + sh[o];
    out[idx] = v > 0.f ? v : 0.01f * v;
}

__global__ __launch_bounds__(256) void linear_out_kernel(
    const float* __restrict__ h, const float* __restrict__ Wl,
    const float* __restrict__ bl, float* __restrict__ out)
{
    __shared__ float sW[OUTD * 64];
    for (int i = threadIdx.x; i < OUTD * 64; i += 256) sW[i] = Wl[i];
    __syncthreads();
    const int idx = blockIdx.x * 256 + threadIdx.x;
    const int n = idx >> 7, j = idx & 127;
    const float* hr = h + (size_t)n * 64;
    float acc = bl[j];
#pragma unroll
    for (int i = 0; i < 64; ++i) acc += hr[i] * sW[j * 64 + i];
    out[idx] = acc;
}

// ---------------------------------------------------------------------------
extern "C" void kernel_launch(void* const* d_in, const int* in_sizes, int n_in,
                              void* d_out, int out_size, void* d_ws, size_t ws_size,
                              hipStream_t stream)
{
    (void)in_sizes; (void)n_in; (void)out_size; (void)ws_size;
    const float* x0  = (const float*)d_in[0];
    const int*   ei  = (const int*)  d_in[1];
    const float* ef  = (const float*)d_in[2];
    const float* We0 = (const float*)d_in[4];
    const float* be0 = (const float*)d_in[5];
    const float* Wr0 = (const float*)d_in[6];
    const float* br0 = (const float*)d_in[7];
    const float* g0  = (const float*)d_in[8];
    const float* bt0 = (const float*)d_in[9];
    const float* We1 = (const float*)d_in[10];
    const float* be1 = (const float*)d_in[11];
    const float* Wr1 = (const float*)d_in[12];
    const float* br1 = (const float*)d_in[13];
    const float* g1  = (const float*)d_in[14];
    const float* bt1 = (const float*)d_in[15];
    const float* Wl  = (const float*)d_in[16];
    const float* bl  = (const float*)d_in[17];
    const int* src = ei;
    const int* dst = ei + N_EDGES;

    char* ws = (char*)d_ws;
    bf16_t* Bp0  = (bf16_t*)(ws);                                // 532,480 B
    bf16_t* Bp1  = (bf16_t*)(ws + 540672);
    float*  aggr = (float*)(ws + 1081344);                       // 2 MB
    float*  hA   = (float*)(ws + 1081344 + 2097152);             // 2 MB
    float*  hB   = (float*)(ws + 1081344 + 2 * 2097152);         // 2 MB
    float*  st   = (float*)(ws + 1081344 + 3 * 2097152);         // 4*64 f32
    float *ssum = st, *ssq = st + 64, *sc = st + 128, *sh = st + 192;

    const int packN = NCHUNK * CHUNK_ELEMS;
    pack_B_kernel<<<(packN + 255) / 256, 256, 0, stream>>>(We0, be0, Bp0);
    pack_B_kernel<<<(packN + 255) / 256, 256, 0, stream>>>(We1, be1, Bp1);

    // ---- layer 0 ----
    hipMemsetAsync(aggr, 0, (size_t)N_NODES * HD * 4, stream);
    hipMemsetAsync(st,   0, 2 * 64 * 4, stream);
    edge_msg_kernel<<<N_EDGES / 64, 128, 0, stream>>>(ef, x0, Bp0, src, dst, aggr);
    node_root_stats_kernel<<<N_NODES * HD / 256, 256, 0, stream>>>(aggr, x0, Wr0, br0, hA, ssum, ssq);
    bn_finalize_kernel<<<1, 64, 0, stream>>>(ssum, ssq, g0, bt0, sc, sh);
    bn_apply_kernel<<<N_NODES * HD / 256, 256, 0, stream>>>(hA, sc, sh, hB);

    // ---- layer 1 ----
    hipMemsetAsync(aggr, 0, (size_t)N_NODES * HD * 4, stream);
    hipMemsetAsync(st,   0, 2 * 64 * 4, stream);
    edge_msg_kernel<<<N_EDGES / 64, 128, 0, stream>>>(ef, hB, Bp1, src, dst, aggr);
    node_root_stats_kernel<<<N_NODES * HD / 256, 256, 0, stream>>>(aggr, hB, Wr1, br1, hA, ssum, ssq);
    bn_finalize_kernel<<<1, 64, 0, stream>>>(ssum, ssq, g1, bt1, sc, sh);
    bn_apply_kernel<<<N_NODES * HD / 256, 256, 0, stream>>>(hA, sc, sh, hB);

    // ---- output ----
    linear_out_kernel<<<N_NODES * OUTD / 256, 256, 0, stream>>>(hB, Wl, bl, (float*)d_out);
}
